// BidirectionalAttention_1262720385052
// MI455X (gfx1250) — matrix operand order repaired and checked by reading, not compiled, not run on hardware
//
#include <hip/hip_runtime.h>
#include <hip/hip_bf16.h>

#define MASK_FILL (-1e-07f)

typedef __attribute__((ext_vector_type(16))) __bf16 bf16x16;
typedef __attribute__((ext_vector_type(8)))  __bf16 bf16x8;
typedef __attribute__((ext_vector_type(8)))  float  f32x8;
typedef __attribute__((ext_vector_type(4)))  float  f32x4;

static constexpr int B_  = 32;
static constexpr int L1_ = 1024;
static constexpr int L2_ = 1024;
static constexpr int D_  = 512;

// ---------------------------------------------------------------- helpers

__device__ __forceinline__ f32x8 zero8() {
  f32x8 z;
#pragma unroll
  for (int e = 0; e < 8; ++e) z[e] = 0.0f;
  return z;
}

__device__ __forceinline__ f32x8 wmma_bf16(bf16x16 a, bf16x16 b, f32x8 c) {
  // D = A(16x32) * B(32x16) + C, fp32 accumulate
  return __builtin_amdgcn_wmma_f32_16x16x32_bf16(false, a, false, b,
                                                 (short)0, c, false, false);
}

// A-fragment (16x32 bf16) from row-major fp32, rows = M, contiguous K.
// Lane L: M = L&15, half h = L>>4; elems e<8 -> K = 8h+e ; e>=8 -> K = 16+8h+(e-8)
__device__ __forceinline__ bf16x16 load_a_f32(const float* row_base, int ld,
                                              int lane, int k0) {
  const int m = lane & 15, h = lane >> 4;
  const float* r = row_base + (size_t)m * ld + k0;
  f32x4 x0 = *(const f32x4*)(r + 8 * h);
  f32x4 x1 = *(const f32x4*)(r + 8 * h + 4);
  f32x4 y0 = *(const f32x4*)(r + 16 + 8 * h);
  f32x4 y1 = *(const f32x4*)(r + 16 + 8 * h + 4);
  bf16x16 a;
#pragma unroll
  for (int j = 0; j < 4; ++j) {
    a[j]      = (__bf16)x0[j];
    a[4 + j]  = (__bf16)x1[j];
    a[8 + j]  = (__bf16)y0[j];
    a[12 + j] = (__bf16)y1[j];
  }
  return a;
}

// B-fragment (32x16) from an [N][K] fp32 array (lane holds column N=L&15,
// elems e<8 -> K = 8h+e ; e>=8 -> K = 16+8h+(e-8)).
__device__ __forceinline__ bf16x16 load_b_f32rows(const float* nk_base, int ld,
                                                  int lane, int k0) {
  const int n = lane & 15, h = lane >> 4;
  const float* r = nk_base + (size_t)n * ld + k0;
  f32x4 x0 = *(const f32x4*)(r + 8 * h);
  f32x4 x1 = *(const f32x4*)(r + 8 * h + 4);
  f32x4 x2 = *(const f32x4*)(r + 16 + 8 * h);
  f32x4 x3 = *(const f32x4*)(r + 16 + 8 * h + 4);
  bf16x16 b;
#pragma unroll
  for (int j = 0; j < 4; ++j) {
    b[j]      = (__bf16)x0[j];
    b[4 + j]  = (__bf16)x1[j];
    b[8 + j]  = (__bf16)x2[j];
    b[12 + j] = (__bf16)x3[j];
  }
  return b;
}

// B-fragment from an [N][K] bf16 array: two 16-byte loads at K = 8h and 16+8h.
__device__ __forceinline__ bf16x16 load_b_bf16(const __bf16* nk_base, int ld,
                                               int lane, int k0) {
  const int n = lane & 15, h = lane >> 4;
  const __bf16* p = nk_base + (size_t)n * ld + k0;
  bf16x8 c0 = *(const bf16x8*)(p + 8 * h);
  bf16x8 c1 = *(const bf16x8*)(p + 16 + 8 * h);
  return __builtin_shufflevector(c0, c1, 0, 1, 2, 3, 4, 5, 6, 7, 8, 9, 10, 11, 12, 13, 14, 15);
}

// ---------------------------------------------------------------- kernel 0
// fp32 [b][R][C] -> bf16 [b][C][R]  (LDS-tiled transpose + convert)
__global__ __launch_bounds__(256) void transpose_to_bf16(
    const float* __restrict__ src, __bf16* __restrict__ dst, int R, int C) {
  __shared__ __bf16 t[32][34];
  const int b  = blockIdx.z;
  const int c0 = blockIdx.x * 32, r0 = blockIdx.y * 32;
  const int tx = threadIdx.x & 31, ty = threadIdx.x >> 5;
  const float* s = src + (size_t)b * R * C;
  __bf16* d = dst + (size_t)b * R * C;
#pragma unroll
  for (int i = 0; i < 32; i += 8)
    t[ty + i][tx] = (__bf16)s[(size_t)(r0 + ty + i) * C + (c0 + tx)];
  __syncthreads();
#pragma unroll
  for (int i = 0; i < 32; i += 8)
    d[(size_t)(c0 + ty + i) * R + (r0 + tx)] = t[tx][ty + i];
}

// ---------------------------------------------------------------- kernel 1
// sim[b][l][m] = sum_d v1[b][l][d] * v2[b][m][d]   (bf16 WMMA, fp32 acc)
__global__ __launch_bounds__(32) void sim_kernel(const float* __restrict__ v1,
                                                 const float* __restrict__ v2,
                                                 float* __restrict__ sim) {
  const int b = blockIdx.z, tl = blockIdx.y, tm = blockIdx.x;
  const int lane = threadIdx.x;
  const float* v1b = v1 + ((size_t)b * L1_ + tl * 64) * D_;
  const float* v2b = v2 + ((size_t)b * L2_ + tm * 64) * D_;
  f32x8 acc[4][4];
#pragma unroll
  for (int i = 0; i < 4; ++i)
#pragma unroll
    for (int j = 0; j < 4; ++j) acc[i][j] = zero8();

  for (int k0 = 0; k0 < D_; k0 += 32) {
    bf16x16 aF[4], bF[4];
#pragma unroll
    for (int i = 0; i < 4; ++i)
      aF[i] = load_a_f32(v1b + (size_t)(16 * i) * D_, D_, lane, k0);
#pragma unroll
    for (int j = 0; j < 4; ++j)
      bF[j] = load_b_f32rows(v2b + (size_t)(16 * j) * D_, D_, lane, k0);
#pragma unroll
    for (int i = 0; i < 4; ++i)
#pragma unroll
      for (int j = 0; j < 4; ++j) acc[i][j] = wmma_bf16(aF[i], bF[j], acc[i][j]);
  }
  const int n = lane & 15, h = lane >> 4;
#pragma unroll
  for (int i = 0; i < 4; ++i)
#pragma unroll
    for (int r = 0; r < 8; ++r) {
      const int row = tl * 64 + 16 * i + 8 * h + r;
      float* orow = sim + ((size_t)b * L1_ + row) * L2_ + tm * 64 + n;
#pragma unroll
      for (int j = 0; j < 4; ++j) orow[16 * j] = acc[i][j][r];
    }
}

// ---------------------------------------------------------------- kernel 2a
// softmax(axis=2) stats per (b,l), masked-fill with v2_mask along m
__global__ __launch_bounds__(256) void row_stats(
    const float* __restrict__ sim, const unsigned char* __restrict__ m2,
    float* __restrict__ rmax, float* __restrict__ rsum) {
  const int warp = threadIdx.x >> 5, lane = threadIdx.x & 31;
  const int row = blockIdx.x * 8 + warp;      // 0 .. B*L1-1
  const int b = row >> 10;
  const float* p = sim + (size_t)row * L2_;
  const unsigned char* mm = m2 + b * L2_;
  float v[32];
  float mx = -3.0e38f;
#pragma unroll
  for (int i = 0; i < 32; ++i) {
    const int mcol = lane + 32 * i;
    float x = p[mcol];
    x = mm[mcol] ? MASK_FILL : x;
    v[i] = x;
    mx = fmaxf(mx, x);
  }
#pragma unroll
  for (int o = 16; o > 0; o >>= 1) mx = fmaxf(mx, __shfl_xor(mx, o, 32));
  float s = 0.0f;
#pragma unroll
  for (int i = 0; i < 32; ++i) s += __expf(v[i] - mx);
#pragma unroll
  for (int o = 16; o > 0; o >>= 1) s += __shfl_xor(s, o, 32);
  if (lane == 0) { rmax[row] = mx; rsum[row] = s; }
}

// ---------------------------------------------------------------- kernel 2b
// softmax(axis=1) stats per (b,m), masked-fill with v1_mask along l
__global__ __launch_bounds__(256) void col_stats(
    const float* __restrict__ sim, const unsigned char* __restrict__ m1,
    float* __restrict__ cmax, float* __restrict__ csum) {
  const int b = blockIdx.y;
  const int m = blockIdx.x * 256 + threadIdx.x;
  const float* p = sim + (size_t)b * L1_ * L2_ + m;
  const unsigned char* mm = m1 + b * L1_;
  float mx = -3.0e38f, s = 0.0f;
  for (int l = 0; l < L1_; ++l) {
    float x = p[(size_t)l * L2_];
    x = mm[l] ? MASK_FILL : x;
    if (x > mx) { s = s * __expf(mx - x) + 1.0f; mx = x; }
    else        { s += __expf(x - mx); }
  }
  cmax[b * L2_ + m] = mx;
  csum[b * L2_ + m] = s;
}

// ---------------------------------------------------------------- kernel 3
// attended_v1[b][l][d] = sum_m softmax_row(sim)[l][m] * v2[m][d]; zero if v1_mask
__global__ __launch_bounds__(32) void av1_kernel(
    const float* __restrict__ sim, const __bf16* __restrict__ v2T,
    const float* __restrict__ rmax, const float* __restrict__ rsum,
    const unsigned char* __restrict__ m1, const unsigned char* __restrict__ m2,
    float* __restrict__ out1) {
  const int b = blockIdx.z, tl = blockIdx.y, td = blockIdx.x;
  const int lane = threadIdx.x;
  const int row0 = tl * 64, col0 = td * 64;
  const int mr = lane & 15, h = lane >> 4;
  float rmx[4], rin[4];
#pragma unroll
  for (int i = 0; i < 4; ++i) {
    const int l = row0 + 16 * i + mr;
    rmx[i] = rmax[b * L1_ + l];
    rin[i] = 1.0f / rsum[b * L1_ + l];
  }
  f32x8 acc[4][4];
#pragma unroll
  for (int i = 0; i < 4; ++i)
#pragma unroll
    for (int j = 0; j < 4; ++j) acc[i][j] = zero8();

  const float* simb = sim + ((size_t)b * L1_ + row0) * L2_;
  const unsigned char* mm2 = m2 + b * L2_;
  const __bf16* v2Tb = v2T + ((size_t)b * D_ + col0) * L2_;

  for (int k0 = 0; k0 < L2_; k0 += 32) {
    const unsigned long long w0 = *(const unsigned long long*)(mm2 + k0 + 8 * h);
    const unsigned long long w1 = *(const unsigned long long*)(mm2 + k0 + 16 + 8 * h);
    bf16x16 aF[4];
#pragma unroll
    for (int i = 0; i < 4; ++i) {
      const float* r = simb + (size_t)(16 * i + mr) * L2_ + k0;
      f32x4 x0 = *(const f32x4*)(r + 8 * h);
      f32x4 x1 = *(const f32x4*)(r + 8 * h + 4);
      f32x4 y0 = *(const f32x4*)(r + 16 + 8 * h);
      f32x4 y1 = *(const f32x4*)(r + 16 + 8 * h + 4);
      bf16x16 a;
#pragma unroll
      for (int j = 0; j < 4; ++j) {
        const float e0 = ((w0 >> (8 * j)) & 1ull)       ? MASK_FILL : x0[j];
        const float e1 = ((w0 >> (8 * (j + 4))) & 1ull) ? MASK_FILL : x1[j];
        const float e2 = ((w1 >> (8 * j)) & 1ull)       ? MASK_FILL : y0[j];
        const float e3 = ((w1 >> (8 * (j + 4))) & 1ull) ? MASK_FILL : y1[j];
        a[j]      = (__bf16)(__expf(e0 - rmx[i]) * rin[i]);
        a[4 + j]  = (__bf16)(__expf(e1 - rmx[i]) * rin[i]);
        a[8 + j]  = (__bf16)(__expf(e2 - rmx[i]) * rin[i]);
        a[12 + j] = (__bf16)(__expf(e3 - rmx[i]) * rin[i]);
      }
      aF[i] = a;
    }
    bf16x16 bF[4];
#pragma unroll
    for (int j = 0; j < 4; ++j)
      bF[j] = load_b_bf16(v2Tb + (size_t)(16 * j) * L2_, L2_, lane, k0);
#pragma unroll
    for (int i = 0; i < 4; ++i)
#pragma unroll
      for (int j = 0; j < 4; ++j) acc[i][j] = wmma_bf16(aF[i], bF[j], acc[i][j]);
  }

  const unsigned char* mm1 = m1 + b * L1_;
  const int n = lane & 15;
#pragma unroll
  for (int i = 0; i < 4; ++i)
#pragma unroll
    for (int r = 0; r < 8; ++r) {
      const int l = row0 + 16 * i + 8 * h + r;
      const float zm = mm1[l] ? 0.0f : 1.0f;
      float* orow = out1 + ((size_t)b * L1_ + l) * D_ + col0 + n;
#pragma unroll
      for (int j = 0; j < 4; ++j) orow[16 * j] = zm * acc[i][j][r];
    }
}

// ---------------------------------------------------------------- kernel 4
// attended_v2[b][m][d] = sum_l softmax_col(sim)[l][m] * v1[l][d]; zero if v2_mask
// A = P_col^T (transposed sim access) -> staged through LDS as bf16 P values.
__global__ __launch_bounds__(32) void av2_kernel(
    const float* __restrict__ sim, const __bf16* __restrict__ v1T,
    const float* __restrict__ cmax, const float* __restrict__ csum,
    const unsigned char* __restrict__ m1, const unsigned char* __restrict__ m2,
    float* __restrict__ out2) {
  const int b = blockIdx.z, tm = blockIdx.y, td = blockIdx.x;
  const int lane = threadIdx.x;
  const int row0 = tm * 64, col0 = td * 64;   // rows = m (L2), cols = d
  __shared__ __bf16 st[64][40];               // [m_local][l_local], 80B row stride

  const int h = lane >> 4;
  const float cm0 = cmax[b * L2_ + row0 + 2 * lane];
  const float cm1 = cmax[b * L2_ + row0 + 2 * lane + 1];
  const float ci0 = 1.0f / csum[b * L2_ + row0 + 2 * lane];
  const float ci1 = 1.0f / csum[b * L2_ + row0 + 2 * lane + 1];
  const unsigned char* mm1 = m1 + b * L1_;
  const __bf16* v1Tb = v1T + ((size_t)b * D_ + col0) * L1_;

  f32x8 acc[4][4];
#pragma unroll
  for (int i = 0; i < 4; ++i)
#pragma unroll
    for (int j = 0; j < 4; ++j) acc[i][j] = zero8();

  for (int k0 = 0; k0 < L1_; k0 += 32) {
    // stage 32(l) x 64(m) block of P_col, transposed, already exp'ed & scaled
    for (int ll = 0; ll < 32; ++ll) {
      const int l = k0 + ll;
      const float2 x =
          *(const float2*)(sim + ((size_t)b * L1_ + l) * L2_ + row0 + 2 * lane);
      const bool mk = mm1[l] != 0;
      const float xa = mk ? MASK_FILL : x.x;
      const float xb = mk ? MASK_FILL : x.y;
      st[2 * lane][ll]     = (__bf16)(__expf(xa - cm0) * ci0);
      st[2 * lane + 1][ll] = (__bf16)(__expf(xb - cm1) * ci1);
    }
    __syncthreads();
    bf16x16 aF[4], bF[4];
#pragma unroll
    for (int i = 0; i < 4; ++i) {
      const int ma = 16 * i + (lane & 15);
      bf16x8 c0 = *(const bf16x8*)&st[ma][8 * h];
      bf16x8 c1 = *(const bf16x8*)&st[ma][16 + 8 * h];
      bf16x16 a;
#pragma unroll
      for (int j = 0; j < 8; ++j) { a[j] = c0[j]; a[8 + j] = c1[j]; }
      aF[i] = a;
    }
#pragma unroll
    for (int j = 0; j < 4; ++j)
      bF[j] = load_b_bf16(v1Tb + (size_t)(16 * j) * L1_, L1_, lane, k0);
#pragma unroll
    for (int i = 0; i < 4; ++i)
#pragma unroll
      for (int j = 0; j < 4; ++j) acc[i][j] = wmma_bf16(aF[i], bF[j], acc[i][j]);
    __syncthreads();
  }

  const unsigned char* mm2 = m2 + b * L2_;
  const int n = lane & 15;
#pragma unroll
  for (int i = 0; i < 4; ++i)
#pragma unroll
    for (int r = 0; r < 8; ++r) {
      const int m = row0 + 16 * i + 8 * h + r;
      const float zm = mm2[m] ? 0.0f : 1.0f;
      float* orow = out2 + ((size_t)b * L2_ + m) * D_ + col0 + n;
#pragma unroll
      for (int j = 0; j < 4; ++j) orow[16 * j] = zm * acc[i][j][r];
    }
}

// ---------------------------------------------------------------- launch

extern "C" void kernel_launch(void* const* d_in, const int* in_sizes, int n_in,
                              void* d_out, int out_size, void* d_ws,
                              size_t ws_size, hipStream_t stream) {
  (void)in_sizes; (void)n_in; (void)out_size; (void)ws_size;
  const float*         v1      = (const float*)d_in[0];
  const unsigned char* v1_mask = (const unsigned char*)d_in[1];  // JAX bool = 1 byte
  const float*         v2      = (const float*)d_in[2];
  const unsigned char* v2_mask = (const unsigned char*)d_in[3];

  float* out1 = (float*)d_out;                       // attended_v1 [B,L1,D]
  float* out2 = out1 + (size_t)B_ * L1_ * D_;        // attended_v2 [B,L2,D]

  // workspace: sim(128MB) + v1T(32MB) + v2T(32MB) + stats(512KB) ~= 192.5MB
  char* ws = (char*)d_ws;
  float* sim = (float*)ws;
  size_t off = (size_t)B_ * L1_ * L2_ * sizeof(float);
  __bf16* v1T = (__bf16*)(ws + off); off += (size_t)B_ * D_ * L1_ * sizeof(__bf16);
  __bf16* v2T = (__bf16*)(ws + off); off += (size_t)B_ * D_ * L2_ * sizeof(__bf16);
  float* rmax = (float*)(ws + off);  off += (size_t)B_ * L1_ * sizeof(float);
  float* rsum = (float*)(ws + off);  off += (size_t)B_ * L1_ * sizeof(float);
  float* cmax = (float*)(ws + off);  off += (size_t)B_ * L2_ * sizeof(float);
  float* csum = (float*)(ws + off);

  transpose_to_bf16<<<dim3(D_ / 32, L1_ / 32, B_), 256, 0, stream>>>(v1, v1T, L1_, D_);
  transpose_to_bf16<<<dim3(D_ / 32, L2_ / 32, B_), 256, 0, stream>>>(v2, v2T, L2_, D_);
  sim_kernel<<<dim3(L2_ / 64, L1_ / 64, B_), 32, 0, stream>>>(v1, v2, sim);
  row_stats<<<dim3(B_ * L1_ / 8), 256, 0, stream>>>(sim, v2_mask, rmax, rsum);
  col_stats<<<dim3(L2_ / 256, B_), 256, 0, stream>>>(sim, v1_mask, cmax, csum);
  av1_kernel<<<dim3(D_ / 64, L1_ / 64, B_), 32, 0, stream>>>(sim, v2T, rmax, rsum,
                                                             v1_mask, v2_mask, out1);
  av2_kernel<<<dim3(D_ / 64, L2_ / 64, B_), 32, 0, stream>>>(sim, v1T, cmax, csum,
                                                             v1_mask, v2_mask, out2);
}
